// GATEncoder_53704271069652
// MI455X (gfx1250) — matrix operand order repaired and checked by reading, not compiled, not run on hardware
//
#include <hip/hip_runtime.h>
#include <cstdint>
#include <cstddef>

#define B_    8
#define XN_   128
#define ON_   64
#define D_    512
#define HID_  2048
#define NH_   8
#define DK_   64
#define OBJ_  2048
#define L_    2
#define XTOK  (B_ * XN_)   /* 1024 */
#define OTOK  (B_ * ON_)   /* 512  */

typedef __attribute__((ext_vector_type(16))) __bf16 bf16x16;
typedef __attribute__((ext_vector_type(8)))  float  floatx8;

// ---------------------------------------------------------------------------
// Weight convert + transpose:  W (K x N, f32, row-major) -> Wt (N x K, bf16)
// ---------------------------------------------------------------------------
__global__ void convT_kernel(const float* __restrict__ W, __bf16* __restrict__ Wt,
                             int K, int N) {
  int idx = blockIdx.x * blockDim.x + threadIdx.x;
  if (idx < K * N) {
    int k = idx / N;
    int n = idx - k * N;
    Wt[(size_t)n * K + k] = (__bf16)W[idx];
  }
}

// ---------------------------------------------------------------------------
// GEMM: C[M,N] = A[M,K](f32, cvt->bf16) x W[K,N] (bf16, stored transposed NxK)
// One wave (32 threads) computes a 32x64 tile of C: 2 A-fragments (rows) x
// 4 B-fragments (cols) -> 8 WMMA per 32-deep k-step, B reused across rows.
// ---------------------------------------------------------------------------
__global__ __launch_bounds__(32)
void gemm_wmma_kernel(const float* __restrict__ A, const __bf16* __restrict__ Wt,
                      const float* __restrict__ bias, float* __restrict__ C,
                      int M, int N, int K, int relu) {
  const int lane = threadIdx.x;
  const int m16  = lane & 15;
  const int half = lane >> 4;
  const int tm   = blockIdx.x << 5;  // 32-row tile
  const int tn   = blockIdx.y << 6;  // 64-col tile

  floatx8 acc[2][4] = {};

  const float*  arow0 = A  + (size_t)(tm + m16) * K;
  const float*  arow1 = A  + (size_t)(tm + 16 + m16) * K;
  const __bf16* wcol  = Wt + (size_t)(tn + m16) * K;

  for (int kk = 0; kk < K; kk += 32) {
    // ---- A fragments: 16x32 bf16 each. Per lane one contiguous 16-run:
    //      k = kk + 16*half + [0..15]
    const int ka = kk + (half << 4);
    bf16x16 af0, af1;
    {
      float4 a0 = *(const float4*)(arow0 + ka);
      float4 a1 = *(const float4*)(arow0 + ka + 4);
      float4 a2 = *(const float4*)(arow0 + ka + 8);
      float4 a3 = *(const float4*)(arow0 + ka + 12);
      af0[0]  = (__bf16)a0.x; af0[1]  = (__bf16)a0.y; af0[2]  = (__bf16)a0.z; af0[3]  = (__bf16)a0.w;
      af0[4]  = (__bf16)a1.x; af0[5]  = (__bf16)a1.y; af0[6]  = (__bf16)a1.z; af0[7]  = (__bf16)a1.w;
      af0[8]  = (__bf16)a2.x; af0[9]  = (__bf16)a2.y; af0[10] = (__bf16)a2.z; af0[11] = (__bf16)a2.w;
      af0[12] = (__bf16)a3.x; af0[13] = (__bf16)a3.y; af0[14] = (__bf16)a3.z; af0[15] = (__bf16)a3.w;
    }
    {
      float4 a0 = *(const float4*)(arow1 + ka);
      float4 a1 = *(const float4*)(arow1 + ka + 4);
      float4 a2 = *(const float4*)(arow1 + ka + 8);
      float4 a3 = *(const float4*)(arow1 + ka + 12);
      af1[0]  = (__bf16)a0.x; af1[1]  = (__bf16)a0.y; af1[2]  = (__bf16)a0.z; af1[3]  = (__bf16)a0.w;
      af1[4]  = (__bf16)a1.x; af1[5]  = (__bf16)a1.y; af1[6]  = (__bf16)a1.z; af1[7]  = (__bf16)a1.w;
      af1[8]  = (__bf16)a2.x; af1[9]  = (__bf16)a2.y; af1[10] = (__bf16)a2.z; af1[11] = (__bf16)a2.w;
      af1[12] = (__bf16)a3.x; af1[13] = (__bf16)a3.y; af1[14] = (__bf16)a3.z; af1[15] = (__bf16)a3.w;
    }

    // ---- B fragments: 32x16 bf16. Per lane-column one contiguous 16-run:
    //      k = kk + 16*half + [0..15]  (32-byte load from transposed weights)
    const int kb = kk + (half << 4);
#pragma unroll
    for (int j = 0; j < 4; ++j) {
      bf16x16 bf = *(const bf16x16*)(wcol + (size_t)(16 * j) * K + kb);
      acc[0][j] = __builtin_amdgcn_wmma_f32_16x16x32_bf16(
          false, af0, false, bf, (short)0, acc[0][j], false, false);
      acc[1][j] = __builtin_amdgcn_wmma_f32_16x16x32_bf16(
          false, af1, false, bf, (short)0, acc[1][j], false, false);
    }
  }

#pragma unroll
  for (int i = 0; i < 2; ++i) {
#pragma unroll
    for (int j = 0; j < 4; ++j) {
      int col  = tn + (j << 4) + m16;
      float bv = bias ? bias[col] : 0.f;
#pragma unroll
      for (int r = 0; r < 8; ++r) {
        int row  = tm + (i << 4) + r + (half << 3);
        float vv = acc[i][j][r] + bv;
        if (relu) vv = fmaxf(vv, 0.f);
        C[(size_t)row * N + col] = vv;
      }
    }
  }
}

// ---------------------------------------------------------------------------
// Self-attention on x set: one block per (b, head, query-row). 128 keys.
// q,k,v: (B*XN, D) with head h in columns [h*DK, (h+1)*DK).
// ---------------------------------------------------------------------------
__global__ __launch_bounds__(128)
void attn_x_kernel(const float* __restrict__ q, const float* __restrict__ k,
                   const float* __restrict__ v, const float* __restrict__ mask,
                   float* __restrict__ out) {
  const int b = blockIdx.z, h = blockIdx.y, qi = blockIdx.x;
  const int tid = threadIdx.x;
  __shared__ float sc[XN_];
  __shared__ float red[XN_];
  const float* qv = q + (size_t)(b * XN_ + qi) * D_ + h * DK_;
  const float* kv = k + (size_t)(b * XN_ + tid) * D_ + h * DK_;
  float s = 0.f;
#pragma unroll 8
  for (int d = 0; d < DK_; ++d) s += qv[d] * kv[d];
  s *= 0.125f;  // 1/sqrt(64)
  if (mask[b * XN_ + tid] <= 0.f) s = -1e9f;
  red[tid] = s; __syncthreads();
  for (int st = 64; st > 0; st >>= 1) {
    if (tid < st) red[tid] = fmaxf(red[tid], red[tid + st]);
    __syncthreads();
  }
  float mx = red[0]; __syncthreads();
  float e = __expf(s - mx);
  sc[tid] = e; red[tid] = e; __syncthreads();
  for (int st = 64; st > 0; st >>= 1) {
    if (tid < st) red[tid] += red[tid + st];
    __syncthreads();
  }
  float inv = 1.f / red[0];
  if (tid < DK_) {
    float acc = 0.f;
    for (int kk = 0; kk < XN_; ++kk)
      acc += sc[kk] * v[(size_t)(b * XN_ + kk) * D_ + h * DK_ + tid];
    out[(size_t)(b * XN_ + qi) * D_ + h * DK_ + tid] = acc * inv;
  }
}

// Self-attention on obj set (64 keys; v is the raw input o, no Wv/Wo).
__global__ __launch_bounds__(64)
void attn_o_kernel(const float* __restrict__ q, const float* __restrict__ k,
                   const float* __restrict__ v, const float* __restrict__ mask,
                   float* __restrict__ out) {
  const int b = blockIdx.z, h = blockIdx.y, qi = blockIdx.x;
  const int tid = threadIdx.x;
  __shared__ float sc[ON_];
  __shared__ float red[ON_];
  const float* qv = q + (size_t)(b * ON_ + qi) * D_ + h * DK_;
  const float* kv = k + (size_t)(b * ON_ + tid) * D_ + h * DK_;
  float s = 0.f;
#pragma unroll 8
  for (int d = 0; d < DK_; ++d) s += qv[d] * kv[d];
  s *= 0.125f;
  if (mask[b * ON_ + tid] <= 0.f) s = -1e9f;
  red[tid] = s; __syncthreads();
  for (int st = 32; st > 0; st >>= 1) {
    if (tid < st) red[tid] = fmaxf(red[tid], red[tid + st]);
    __syncthreads();
  }
  float mx = red[0]; __syncthreads();
  float e = __expf(s - mx);
  sc[tid] = e; red[tid] = e; __syncthreads();
  for (int st = 32; st > 0; st >>= 1) {
    if (tid < st) red[tid] += red[tid + st];
    __syncthreads();
  }
  float inv = 1.f / red[0];
  float acc = 0.f;
  for (int kk = 0; kk < ON_; ++kk)
    acc += sc[kk] * v[(size_t)(b * ON_ + kk) * D_ + h * DK_ + tid];
  out[(size_t)(b * ON_ + qi) * D_ + h * DK_ + tid] = acc * inv;
}

// ---------------------------------------------------------------------------
// out = LayerNorm(res + y) * gamma + beta, row width D_=512, block 256.
// ---------------------------------------------------------------------------
__global__ __launch_bounds__(256)
void add_ln_kernel(const float* __restrict__ res, const float* __restrict__ y,
                   const float* __restrict__ gamma, const float* __restrict__ beta,
                   float* __restrict__ out) {
  const int row = blockIdx.x, tid = threadIdx.x;
  const float* r0 = res + (size_t)row * D_;
  const float* y0 = y + (size_t)row * D_;
  float t0 = r0[tid] + y0[tid];
  float t1 = r0[tid + 256] + y0[tid + 256];
  __shared__ float sb[256];
  sb[tid] = t0 + t1; __syncthreads();
  for (int st = 128; st > 0; st >>= 1) { if (tid < st) sb[tid] += sb[tid + st]; __syncthreads(); }
  float mean = sb[0] * (1.f / (float)D_); __syncthreads();
  float d0 = t0 - mean, d1 = t1 - mean;
  sb[tid] = d0 * d0 + d1 * d1; __syncthreads();
  for (int st = 128; st > 0; st >>= 1) { if (tid < st) sb[tid] += sb[tid + st]; __syncthreads(); }
  float inv = rsqrtf(sb[0] * (1.f / (float)D_) + 1e-5f);
  out[(size_t)row * D_ + tid]       = gamma[tid]       * d0 * inv + beta[tid];
  out[(size_t)row * D_ + tid + 256] = gamma[tid + 256] * d1 * inv + beta[tid + 256];
}

// ---------------------------------------------------------------------------
// x2o gate: out[b,x,d] = sum_o sigmoid(gx[b,x,d]+go[b,o,d]+bias[d]) * M[b,x,o] * newo[b,o,d]
// M[b,x,o] is uniform across the block -> scalar branch skips ~70% zeros.
// ---------------------------------------------------------------------------
__global__ __launch_bounds__(256)
void x2o_gate_kernel(const float* __restrict__ gx, const float* __restrict__ go,
                     const float* __restrict__ bias, const float* __restrict__ matrix,
                     const float* __restrict__ newo, float* __restrict__ out) {
  const int b = blockIdx.y, xr = blockIdx.x, tid = threadIdx.x;
  const int d0 = tid, d1 = tid + 256;
  const size_t xbase = (size_t)(b * XN_ + xr) * D_;
  float g0 = gx[xbase + d0] + bias[d0];
  float g1 = gx[xbase + d1] + bias[d1];
  const float* mrow = matrix + (size_t)(b * XN_ + xr) * ON_;
  float acc0 = 0.f, acc1 = 0.f;
  for (int o = 0; o < ON_; ++o) {
    float m = mrow[o];
    if (m != 0.f) {
      size_t ob = (size_t)(b * ON_ + o) * D_;
      float t0 = g0 + go[ob + d0];
      float t1 = g1 + go[ob + d1];
      acc0 += m * newo[ob + d0] / (1.f + __expf(-t0));
      acc1 += m * newo[ob + d1] / (1.f + __expf(-t1));
    }
  }
  out[xbase + d0] = acc0;
  out[xbase + d1] = acc1;
}

// o2x gate: out[b,o,d] = sum_x sigmoid(hx[b,x,d]+ho[b,o,d]+bias[d]) * M[b,x,o] * newx[b,x,d]
__global__ __launch_bounds__(256)
void o2x_gate_kernel(const float* __restrict__ hx, const float* __restrict__ ho,
                     const float* __restrict__ bias, const float* __restrict__ matrix,
                     const float* __restrict__ newx, float* __restrict__ out) {
  const int b = blockIdx.y, orow = blockIdx.x, tid = threadIdx.x;
  const int d0 = tid, d1 = tid + 256;
  const size_t obase = (size_t)(b * ON_ + orow) * D_;
  float g0 = ho[obase + d0] + bias[d0];
  float g1 = ho[obase + d1] + bias[d1];
  float acc0 = 0.f, acc1 = 0.f;
  for (int x = 0; x < XN_; ++x) {
    float m = matrix[(size_t)(b * XN_ + x) * ON_ + orow];
    if (m != 0.f) {
      size_t xb = (size_t)(b * XN_ + x) * D_;
      float t0 = g0 + hx[xb + d0];
      float t1 = g1 + hx[xb + d1];
      acc0 += m * newx[xb + d0] / (1.f + __expf(-t0));
      acc1 += m * newx[xb + d1] / (1.f + __expf(-t1));
    }
  }
  out[obase + d0] = acc0;
  out[obase + d1] = acc1;
}

__global__ void copy_kernel(const float* __restrict__ s, float* __restrict__ d, int n) {
  int i = blockIdx.x * blockDim.x + threadIdx.x;
  if (i < n) d[i] = s[i];
}

// ---------------------------------------------------------------------------
extern "C" void kernel_launch(void* const* d_in, const int* in_sizes, int n_in,
                              void* d_out, int out_size, void* d_ws, size_t ws_size,
                              hipStream_t stream) {
  (void)in_sizes; (void)n_in; (void)out_size; (void)ws_size;
  auto F = [&](int i) { return (const float*)d_in[i]; };
  const float* x_in   = F(0);
  const float* mask   = F(1);
  const float* obj    = F(2);
  const float* omask  = F(3);
  const float* matrix = F(4);

  char* bump = (char*)d_ws;
  auto alloc = [&](size_t bytes) -> void* {
    void* p = bump; bump += (bytes + 255) & ~(size_t)255; return p;
  };
  auto fbuf = [&](size_t n) { return (float*)alloc(n * sizeof(float)); };
  auto bbuf = [&](size_t n) { return (__bf16*)alloc(n * sizeof(__bf16)); };

  // ---- bf16 transposed weights -------------------------------------------
  __bf16* to_w1t = bbuf((size_t)OBJ_ * D_);
  __bf16* to_w2t = bbuf((size_t)D_ * D_);
  __bf16 *xwq[L_], *xwk[L_], *xwv[L_], *xwo[L_], *owq[L_], *owk[L_];
  __bf16 *x2oA[L_], *x2oB[L_], *o2xA[L_], *o2xB[L_];
  __bf16 *fxw1[L_], *fxw2[L_], *fow1[L_], *fow2[L_];
  for (int i = 0; i < L_; ++i) {
    xwq[i] = bbuf((size_t)D_ * D_); xwk[i] = bbuf((size_t)D_ * D_);
    xwv[i] = bbuf((size_t)D_ * D_); xwo[i] = bbuf((size_t)D_ * D_);
    owq[i] = bbuf((size_t)D_ * D_); owk[i] = bbuf((size_t)D_ * D_);
    x2oA[i] = bbuf((size_t)D_ * D_); x2oB[i] = bbuf((size_t)D_ * D_);
    o2xA[i] = bbuf((size_t)D_ * D_); o2xB[i] = bbuf((size_t)D_ * D_);
    fxw1[i] = bbuf((size_t)D_ * HID_); fxw2[i] = bbuf((size_t)HID_ * D_);
    fow1[i] = bbuf((size_t)D_ * HID_); fow2[i] = bbuf((size_t)HID_ * D_);
  }

  // ---- fp32 activation buffers -------------------------------------------
  float* o0    = fbuf((size_t)OTOK * D_);
  float* obuf  = fbuf((size_t)OTOK * D_);
  float* qo    = fbuf((size_t)OTOK * D_);
  float* ko    = fbuf((size_t)OTOK * D_);
  float* ao    = fbuf((size_t)OTOK * D_);
  float* newo  = fbuf((size_t)OTOK * D_);
  float* go_   = fbuf((size_t)OTOK * D_);
  float* ho_   = fbuf((size_t)OTOK * D_);
  float* o2x   = fbuf((size_t)OTOK * D_);
  float* newo2 = fbuf((size_t)OTOK * D_);
  float* fo    = fbuf((size_t)OTOK * D_);
  float* hbo   = fbuf((size_t)OTOK * HID_);
  float* xbuf  = fbuf((size_t)XTOK * D_);
  float* qx    = fbuf((size_t)XTOK * D_);
  float* kx    = fbuf((size_t)XTOK * D_);
  float* vx    = fbuf((size_t)XTOK * D_);
  float* ax    = fbuf((size_t)XTOK * D_);
  float* px    = fbuf((size_t)XTOK * D_);
  float* newx  = fbuf((size_t)XTOK * D_);
  float* gx_   = fbuf((size_t)XTOK * D_);
  float* hx_   = fbuf((size_t)XTOK * D_);
  float* x2o   = fbuf((size_t)XTOK * D_);
  float* newx2 = fbuf((size_t)XTOK * D_);
  float* fx    = fbuf((size_t)XTOK * D_);
  float* hbx   = fbuf((size_t)XTOK * HID_);

  auto convT = [&](const float* W, __bf16* Wt, int K, int N) {
    int total = K * N;
    convT_kernel<<<dim3((total + 255) / 256), dim3(256), 0, stream>>>(W, Wt, K, N);
  };
  auto gemm = [&](const float* A, const __bf16* Wt, const float* bias, float* C,
                  int M, int N, int K, int relu) {
    gemm_wmma_kernel<<<dim3(M / 32, N / 64), dim3(32), 0, stream>>>(A, Wt, bias, C, M, N, K, relu);
  };
  auto addln = [&](const float* res, const float* y, const float* g, const float* be,
                   float* out, int rows) {
    add_ln_kernel<<<dim3(rows), dim3(256), 0, stream>>>(res, y, g, be, out);
  };

  // ---- convert all weights (deterministic, every call) -------------------
  convT(F(5), to_w1t, OBJ_, D_);
  convT(F(7), to_w2t, D_, D_);
  for (int i = 0; i < L_; ++i) {
    size_t wdd = (size_t)i * D_ * D_;
    convT(F(9)  + wdd, xwq[i], D_, D_);
    convT(F(11) + wdd, xwk[i], D_, D_);
    convT(F(13) + wdd, xwv[i], D_, D_);
    convT(F(15) + wdd, xwo[i], D_, D_);
    convT(F(17) + wdd, owq[i], D_, D_);
    convT(F(19) + wdd, owk[i], D_, D_);
    const float* x2ow = F(21) + (size_t)i * 2 * D_ * D_;
    convT(x2ow, x2oA[i], D_, D_);
    convT(x2ow + (size_t)D_ * D_, x2oB[i], D_, D_);
    const float* o2xw = F(23) + (size_t)i * 2 * D_ * D_;
    convT(o2xw, o2xA[i], D_, D_);
    convT(o2xw + (size_t)D_ * D_, o2xB[i], D_, D_);
    convT(F(25) + (size_t)i * D_ * HID_, fxw1[i], D_, HID_);
    convT(F(27) + (size_t)i * HID_ * D_, fxw2[i], HID_, D_);
    convT(F(29) + (size_t)i * D_ * HID_, fow1[i], D_, HID_);
    convT(F(31) + (size_t)i * HID_ * D_, fow2[i], HID_, D_);
  }

  // ---- trans_obj: relu(relu(obj @ W1 + b1) @ W2 + b2) --------------------
  gemm(obj, to_w1t, F(6), o0, OTOK, D_, OBJ_, 1);
  gemm(o0, to_w2t, F(8), obuf, OTOK, D_, D_, 1);

  const float* xcur = x_in;
  for (int i = 0; i < L_; ++i) {
    size_t bd = (size_t)i * D_;
    // x self-attention + LN
    gemm(xcur, xwq[i], F(10) + bd, qx, XTOK, D_, D_, 0);
    gemm(xcur, xwk[i], F(12) + bd, kx, XTOK, D_, D_, 0);
    gemm(xcur, xwv[i], F(14) + bd, vx, XTOK, D_, D_, 0);
    attn_x_kernel<<<dim3(XN_, NH_, B_), dim3(128), 0, stream>>>(qx, kx, vx, mask, ax);
    gemm(ax, xwo[i], F(16) + bd, px, XTOK, D_, D_, 0);
    addln(xcur, px, F(33) + bd, F(34) + bd, newx, XTOK);
    // o self-attention (v = o, no output proj) + LN
    gemm(obuf, owq[i], F(18) + bd, qo, OTOK, D_, D_, 0);
    gemm(obuf, owk[i], F(20) + bd, ko, OTOK, D_, D_, 0);
    attn_o_kernel<<<dim3(ON_, NH_, B_), dim3(64), 0, stream>>>(qo, ko, obuf, omask, ao);
    addln(obuf, ao, F(35) + bd, F(36) + bd, newo, OTOK);
    // cross-set gating
    gemm(newx, x2oA[i], nullptr, gx_, XTOK, D_, D_, 0);
    gemm(newo, x2oB[i], nullptr, go_, OTOK, D_, D_, 0);
    x2o_gate_kernel<<<dim3(XN_, B_), dim3(256), 0, stream>>>(gx_, go_, F(22) + bd, matrix, newo, x2o);
    gemm(newx, o2xA[i], nullptr, hx_, XTOK, D_, D_, 0);
    gemm(newo, o2xB[i], nullptr, ho_, OTOK, D_, D_, 0);
    o2x_gate_kernel<<<dim3(ON_, B_), dim3(256), 0, stream>>>(hx_, ho_, F(24) + bd, matrix, newx, o2x);
    addln(newx, x2o, F(37) + bd, F(38) + bd, newx2, XTOK);
    addln(newo, o2x, F(39) + bd, F(40) + bd, newo2, OTOK);
    // FFNs + LN
    gemm(newx2, fxw1[i], F(26) + (size_t)i * HID_, hbx, XTOK, HID_, D_, 1);
    gemm(hbx, fxw2[i], F(28) + bd, fx, XTOK, D_, HID_, 0);
    addln(newx2, fx, F(41) + bd, F(42) + bd, xbuf, XTOK);
    gemm(newo2, fow1[i], F(30) + (size_t)i * HID_, hbo, OTOK, HID_, D_, 1);
    gemm(hbo, fow2[i], F(32) + bd, fo, OTOK, D_, HID_, 0);
    addln(newo2, fo, F(43) + bd, F(44) + bd, obuf, OTOK);
    xcur = xbuf;
  }

  const int nx = XTOK * D_, no = OTOK * D_;
  copy_kernel<<<dim3((nx + 255) / 256), dim3(256), 0, stream>>>(xbuf, (float*)d_out, nx);
  copy_kernel<<<dim3((no + 255) / 256), dim3(256), 0, stream>>>(obuf, (float*)d_out + nx, no);
}
